// LinearAttention_18975165514222
// MI455X (gfx1250) — matrix operand order repaired and checked by reading, not compiled, not run on hardware
//
#include <hip/hip_runtime.h>
#include <stdint.h>

// ---------------------------------------------------------------------------
// Causal linear attention (quadratic form) for MI455X / gfx1250.
// qf = relu(Q @ Wq), kf = relu(K @ Wk)  (bf16 in ws), V pre-transposed per tile.
// out[m] = sum_{n<=m} (qf[m].kf[n]) V[n] / (sum_{n<=m} qf[m].kf[n] + eps)
// Streaming over key tiles with double-buffered LDS: TDM (tensor_load_to_lds)
// + global_load_async_to_lds_b128 prefetch tile j+1 while WMMAs chew tile j.
// ---------------------------------------------------------------------------

typedef __attribute__((ext_vector_type(16))) __bf16   v16bf;
typedef __attribute__((ext_vector_type(8)))  float    v8f;
typedef __attribute__((ext_vector_type(4)))  uint32_t u32x4;
typedef __attribute__((ext_vector_type(8)))  int      i32x8;
typedef __attribute__((ext_vector_type(4)))  int      i32x4;

#define PITCH 72          // LDS row pitch in bf16 elems (144B: conflict-free, 16B aligned)
#define EPS   1e-6f
#define BB    2
#define HH    16
#define LL    2048
#define DD    64

// Native f32 -> bf16 (hardware cvt, RNE)
__device__ __forceinline__ __bf16 f2bf(float f) { return (__bf16)f; }

// Build a v16bf operand from two 16-byte chunks (elements 0..7 from p0, 8..15 from p1).
__device__ __forceinline__ v16bf ld2x16(const __bf16* p0, const __bf16* p1) {
  union { v16bf v; uint4 q[2]; } u;
  u.q[0] = *reinterpret_cast<const uint4*>(p0);
  u.q[1] = *reinterpret_cast<const uint4*>(p1);
  return u.v;
}

__device__ __forceinline__ v8f wmma_bf16(v16bf a, v16bf b, v8f c) {
  return __builtin_amdgcn_wmma_f32_16x16x32_bf16(false, a, false, b, (short)0, c,
                                                 false, false);
}

// ---- TDM: DMA one 64x64 bf16 row-major tile global -> LDS, padding each
//      128B row with 16B so the LDS destination has row pitch 72 elems. ------
__device__ __forceinline__ void tdm_load_tile_padded(uint32_t lds_byte_off,
                                                     const void* gaddr) {
  const uint64_t ga = (uint64_t)(uintptr_t)gaddr;
  u32x4 g0;
  g0[0] = 1u;                                  // count=1 (valid user descriptor)
  g0[1] = lds_byte_off;                        // lds_addr
  g0[2] = (uint32_t)ga;                        // global_addr[31:0]
  g0[3] = (uint32_t)(ga >> 32) | (2u << 30);   // global_addr[56:32] | type=2
  i32x8 g1;
  // data_size=2B (1<<16) | pad_enable (1<<20) | pad_interval: 32 DW (4<<22)
  // | pad_amount: 4 DW (3<<25)   -> 64 bf16 row then +16B pad = pitch 72
  g1[0] = (1 << 16) | (1 << 20) | (4 << 22) | (3 << 25);
  g1[1] = (64 << 16);                          // tensor_dim0 = 64
  g1[2] = (64 << 16);                          // tensor_dim1 = 64
  g1[3] = (64 << 16);                          // tile_dim0   = 64
  g1[4] = 64;                                  // tile_dim1   = 64
  g1[5] = 64;                                  // tensor_dim0_stride = 64
  g1[6] = 0;
  g1[7] = 0;
  i32x4 z4 = {};
#if defined(__clang_major__) && (__clang_major__ >= 23)
  i32x8 z8 = {};
  __builtin_amdgcn_tensor_load_to_lds(g0, g1, z4, z4, z8, 0);
#else
  __builtin_amdgcn_tensor_load_to_lds(g0, g1, z4, z4, 0);
#endif
}

// ---- Async copy of a 64x64 bf16 row-major tile into LDS (pitch 72). -------
// Each of 128 threads moves 4 x b128; completion via s_wait_asynccnt.
__device__ __forceinline__ void async_fill_tile(__bf16* dst, const __bf16* src,
                                                int tid) {
  const int r = tid >> 1, c0 = (tid & 1) * 32;
  const uint32_t loff = (uint32_t)(uintptr_t)(dst + r * PITCH + c0);
  const int32_t  goff = (r * 64 + c0) * 2;
  const void* base = (const void*)src;
#pragma unroll
  for (int q = 0; q < 4; ++q) {
    asm volatile("global_load_async_to_lds_b128 %0, %1, %2 offset:0"
                 :: "v"(loff + 16u * q), "v"(goff + 16 * q), "s"(base)
                 : "memory");
  }
}

// ---------------------------------------------------------------------------
// Kernel 1: qf = relu(Q@Wq), kf = relu(K@Wk); V -> bf16 transposed per tile.
// Grid: (L/64, B*H), block 128 (4 waves).
// ---------------------------------------------------------------------------
__global__ __launch_bounds__(128)
void lin_featmap_kernel(const float* __restrict__ Q, const float* __restrict__ Kst,
                        const float* __restrict__ V,
                        const float* __restrict__ Wq, const float* __restrict__ Wk,
                        __bf16* __restrict__ qf, __bf16* __restrict__ kf,
                        __bf16* __restrict__ vb) {
  const int tile = blockIdx.x, bh = blockIdx.y, h = bh % HH;
  const int tid = threadIdx.x, lane = tid & 31, wave = tid >> 5;
  const int half = lane >> 4, lm = lane & 15;

  __shared__ __bf16 sWq[DD * PITCH];   // W transposed: sW[e][d]
  __shared__ __bf16 sWk[DD * PITCH];
  __shared__ __bf16 sVt[64 * PITCH];   // V tile transposed: sVt[d][k]

  const int base = bh * LL * DD;

  {  // stage weights transposed + bf16
    const int d = tid >> 1, c0 = (tid & 1) * 32;
    const float* wq = Wq + h * DD * DD + d * DD + c0;
    const float* wk = Wk + h * DD * DD + d * DD + c0;
#pragma unroll
    for (int t = 0; t < 32; ++t) {
      sWq[(c0 + t) * PITCH + d] = f2bf(wq[t]);
      sWk[(c0 + t) * PITCH + d] = f2bf(wk[t]);
    }
    // V tile: load rows (coalesced f32), scatter transposed into LDS
    const float* vr = V + base + (tile * 64 + d) * DD + c0;
#pragma unroll
    for (int t = 0; t < 32; ++t) sVt[(c0 + t) * PITCH + d] = f2bf(vr[t]);
  }
  __syncthreads();

  const int row  = tile * 64 + wave * 16 + lm;        // A-operand row for this lane
  const int orow = tile * 64 + wave * 16 + half * 8;  // C-layout row base

#pragma unroll
  for (int which = 0; which < 2; ++which) {
    const float*  X  = which ? Kst : Q;
    const __bf16* sW = which ? sWk : sWq;
    __bf16*       Yo = which ? kf  : qf;

    // A layout: lanes 0-15 rows, elems 0..7 = K(half*8..), elems 8..15 = K(16+half*8..).
    v16bf a[2];
#pragma unroll
    for (int ks = 0; ks < 2; ++ks) {
      const float* p0 = X + base + row * DD + ks * 32 + half * 8;
      const float* p1 = p0 + 16;
      v16bf av;
#pragma unroll
      for (int t = 0; t < 8; ++t) { av[t] = f2bf(p0[t]); av[8 + t] = f2bf(p1[t]); }
      a[ks] = av;
    }
#pragma unroll
    for (int nt = 0; nt < 4; ++nt) {
      v8f c = {};
#pragma unroll
      for (int ks = 0; ks < 2; ++ks) {
        // B layout: lane -> column n = nt*16+lm, K(half*8..) and K(16+half*8..) from row n of W^T.
        const __bf16* bp = sW + (nt * 16 + lm) * PITCH + ks * 32 + half * 8;
        c = wmma_bf16(a[ks], ld2x16(bp, bp + 16), c);
      }
#pragma unroll
      for (int r = 0; r < 8; ++r) {
        float v = c[r];
        v = v > 0.0f ? v : 0.0f;                      // ReLU
        Yo[base + (orow + r) * DD + nt * 16 + lm] = f2bf(v);
      }
    }
  }

  {  // write transposed V tile out coalesced: vb[bh][tile][d][k]
    const int d = tid >> 1, c0 = (tid & 1) * 32;
    const uint4* s4 = reinterpret_cast<const uint4*>(&sVt[d * PITCH + c0]);
    uint4* g4 = reinterpret_cast<uint4*>(vb + base + tile * 64 * DD + d * DD + c0);
#pragma unroll
    for (int t = 0; t < 4; ++t) g4[t] = s4[t];
  }
}

// ---------------------------------------------------------------------------
// Kernel 2: streaming causal attention, double-buffered DMA pipeline.
// Grid: (L/64, B*H), block 128. Wave w owns query rows [tile*64+w*16, +16).
// ---------------------------------------------------------------------------
__global__ __launch_bounds__(128)
void lin_attn_kernel(const __bf16* __restrict__ qf, const __bf16* __restrict__ kf,
                     const __bf16* __restrict__ vb, float* __restrict__ out) {
  const int ti = blockIdx.x, bh = blockIdx.y;
  const int tid = threadIdx.x, lane = tid & 31, wave = tid >> 5;
  const int half = lane >> 4, lm = lane & 15;

  __shared__ __bf16 sK[2][64 * PITCH];     // key-feature tile, row-major [n][d]
  __shared__ __bf16 sV[2][64 * PITCH];     // V tile (pre-transposed) [d][k]
  __shared__ __bf16 sS[4][16 * PITCH];     // per-wave score staging (C->A relayout)

  const int base = bh * LL * DD;

  // Q-feature A operands, constant across the key loop.
  v16bf aq[2];
  {
    const __bf16* qr = qf + base + (ti * 64 + wave * 16 + lm) * DD;
#pragma unroll
    for (int ks = 0; ks < 2; ++ks) {
      const __bf16* p = qr + ks * 32 + half * 8;
      aq[ks] = ld2x16(p, p + 16);
    }
  }

  v8f   acc[4]  = {};   // numerator accumulators, 4 d-subtiles of 16
  float dsum[8] = {};   // per-lane partial row sums (denominator)

  // Prologue: start DMA of tile j=0 into buffer 0.
  if (wave == 0)
    tdm_load_tile_padded((uint32_t)(uintptr_t)sK[0], kf + base);
  async_fill_tile(sV[0], vb + base, tid);

  for (int j = 0; j <= ti; ++j) {
    const int cur = j & 1;

    // Own DMA issues for buf[cur] complete; barrier publishes them and also
    // proves every wave finished reading buf[cur^1] in the previous iteration.
    asm volatile("s_wait_asynccnt 0x0" ::: "memory");
    __builtin_amdgcn_s_wait_tensorcnt(0);
    __syncthreads();

    // Prefetch tile j+1 into the other buffer while we compute on this one.
    if (j < ti) {
      if (wave == 0)
        tdm_load_tile_padded((uint32_t)(uintptr_t)sK[cur ^ 1],
                             kf + base + (j + 1) * 64 * DD);
      async_fill_tile(sV[cur ^ 1], vb + base + (j + 1) * 64 * DD, tid);
    }

    const __bf16* cK = sK[cur];
    const __bf16* cV = sV[cur];

    // S = qf_tile @ kf_j^T   (B column n = kf row n -> row-major LDS reads)
    v8f s[4];
#pragma unroll
    for (int nt = 0; nt < 4; ++nt) {
      v8f c = {};
#pragma unroll
      for (int ks = 0; ks < 2; ++ks) {
        const __bf16* bp = &cK[(nt * 16 + lm) * PITCH + ks * 32 + half * 8];
        c = wmma_bf16(aq[ks], ld2x16(bp, bp + 16), c);
      }
      s[nt] = c;
    }

    // Causal mask on the diagonal tile (C layout: elem r -> row r + half*8).
    if (j == ti) {
      const int mb = wave * 16 + half * 8;
#pragma unroll
      for (int nt = 0; nt < 4; ++nt) {
        const int nc = nt * 16 + lm;
#pragma unroll
        for (int r = 0; r < 8; ++r)
          if (nc > mb + r) s[nt][r] = 0.0f;
      }
    }

    // Row-sum partials + stage S (bf16) for the C->A layout round-trip.
    __bf16* sw = sS[wave];
#pragma unroll
    for (int nt = 0; nt < 4; ++nt) {
#pragma unroll
      for (int r = 0; r < 8; ++r) {
        dsum[r] += s[nt][r];
        sw[(half * 8 + r) * PITCH + nt * 16 + lm] = f2bf(s[nt][r]);
      }
    }

    // Re-read S as A operands (intra-wave LDS RAW: DS ops are in-order per wave).
    v16bf as[2];
    {
      const __bf16* sr = sw + lm * PITCH;
#pragma unroll
      for (int ks = 0; ks < 2; ++ks) {
        const __bf16* p = sr + ks * 32 + half * 8;
        as[ks] = ld2x16(p, p + 16);
      }
    }

    // num += S @ V_j  (B column d = row d of transposed V -> contiguous reads)
#pragma unroll
    for (int nt = 0; nt < 4; ++nt) {
#pragma unroll
      for (int ks = 0; ks < 2; ++ks) {
        const __bf16* bp = &cV[(nt * 16 + lm) * PITCH + ks * 32 + half * 8];
        acc[nt] = wmma_bf16(as[ks], ld2x16(bp, bp + 16), acc[nt]);
      }
    }
  }

  // Denominator: reduce each 16-lane half (one half = one set of rows).
#pragma unroll
  for (int r = 0; r < 8; ++r) {
    float v = dsum[r];
    v += __shfl_xor(v, 1, 32);
    v += __shfl_xor(v, 2, 32);
    v += __shfl_xor(v, 4, 32);
    v += __shfl_xor(v, 8, 32);
    dsum[r] = v + EPS;
  }

  const int ob = base + (ti * 64 + wave * 16 + half * 8) * DD;
#pragma unroll
  for (int nt = 0; nt < 4; ++nt)
#pragma unroll
    for (int r = 0; r < 8; ++r)
      out[ob + r * DD + nt * 16 + lm] = acc[nt][r] / dsum[r];
}

// ---------------------------------------------------------------------------
extern "C" void kernel_launch(void* const* d_in, const int* in_sizes, int n_in,
                              void* d_out, int out_size, void* d_ws, size_t ws_size,
                              hipStream_t stream) {
  (void)in_sizes; (void)n_in; (void)out_size; (void)ws_size;
  const float* Q  = (const float*)d_in[0];
  const float* K  = (const float*)d_in[1];
  const float* V  = (const float*)d_in[2];
  const float* Wq = (const float*)d_in[3];
  const float* Wk = (const float*)d_in[4];

  const int N = BB * HH * LL * DD;            // 4,194,304 elements
  __bf16* qf = (__bf16*)d_ws;                 // 8 MB
  __bf16* kf = qf + N;                        // 8 MB
  __bf16* vb = kf + N;                        // 8 MB  (24 MB scratch total)

  dim3 grid(LL / 64, BB * HH);
  lin_featmap_kernel<<<grid, 128, 0, stream>>>(Q, K, V, Wq, Wk, qf, kf, vb);
  lin_attn_kernel<<<grid, 128, 0, stream>>>(qf, kf, vb, (float*)d_out);
}
